// OctonionFusedLinear_28484223108017
// MI455X (gfx1250) — matrix operand order repaired and checked by reading, not compiled, not run on hardware
//
#include <hip/hip_runtime.h>
#include <hip/hip_bf16.h>
#include <stdint.h>

// ---------------------------------------------------------------------------
// Octonion fused linear:  Y[8192 x 4096] = X[8192 x 4096] * Wbig^T
// Wbig[i*512+n][j*512+k] = SIGN[i][j] * W[WIDX[i][j]][n][k]
// Strategy: bf16 WMMA (v_wmma_f32_16x16x32_bf16) with f32 accumulation.
// ---------------------------------------------------------------------------

typedef __attribute__((ext_vector_type(16))) __bf16 v16bf;
typedef __attribute__((ext_vector_type(8)))  float  v8f;

#define BT    8192   // B*T rows of X
#define DIN   4096   // input features  (K total)
#define DOUT  4096   // output features (N total)
#define NC    512    // per-component N
#define KC    512    // per-component K

__constant__ float c_sign[64] = {
    +1,-1,-1,-1,-1,-1,-1,-1,
    +1,+1,+1,-1,+1,-1,-1,+1,
    +1,-1,+1,+1,+1,+1,-1,-1,
    +1,+1,-1,+1,+1,-1,+1,-1,
    +1,-1,-1,-1,+1,+1,+1,+1,
    +1,+1,-1,+1,-1,+1,-1,+1,
    +1,+1,+1,-1,-1,+1,+1,-1,
    +1,-1,+1,+1,-1,-1,+1,+1,
};
__constant__ int c_widx[64] = {
    0,1,2,3,4,5,6,7,
    1,0,3,2,5,4,7,6,
    2,3,0,1,6,7,4,5,
    3,2,1,0,7,6,5,4,
    4,5,6,7,0,1,2,3,
    5,4,7,6,1,0,3,2,
    6,7,4,5,2,3,0,1,
    7,6,5,4,3,2,1,0,
};

__device__ __forceinline__ uint16_t f2bf(float f) {
    // round-to-nearest-even fp32 -> bf16
    uint32_t u = __float_as_uint(f);
    uint32_t r = u + 0x7FFFu + ((u >> 16) & 1u);
    return (uint16_t)(r >> 16);
}

// --------------------------- x: fp32 -> bf16 -------------------------------
// one thread converts 8 consecutive elements (2x float4 load, 1x uint4 store)
__global__ __launch_bounds__(256) void oct_cvt_x(const float* __restrict__ x,
                                                 uint16_t* __restrict__ xb) {
    size_t t = (size_t)blockIdx.x * 256 + threadIdx.x;   // 0 .. BT*DIN/8-1
    const float4* src = (const float4*)x + t * 2;
    float4 a = src[0];
    float4 b = src[1];
    union { uint16_t h[8]; uint4 q; } o;
    o.h[0] = f2bf(a.x); o.h[1] = f2bf(a.y);
    o.h[2] = f2bf(a.z); o.h[3] = f2bf(a.w);
    o.h[4] = f2bf(b.x); o.h[5] = f2bf(b.y);
    o.h[6] = f2bf(b.z); o.h[7] = f2bf(b.w);
    ((uint4*)xb)[t] = o.q;
}

// --------------- build effective weight: gather + sign + bf16 --------------
// Wbig row r = i*512+n, col c = j*512+k ; one thread does 8 consecutive k
__global__ __launch_bounds__(256) void oct_build_w(const float* __restrict__ w,
                                                   uint16_t* __restrict__ wb) {
    size_t t   = (size_t)blockIdx.x * 256 + threadIdx.x; // 0 .. 4096*512-1
    int    row = (int)(t >> 9);          // 0..4095
    int    c8  = ((int)t & 511) << 3;    // col start, multiple of 8
    int    i   = row >> 9, n = row & (NC - 1);
    int    j   = c8 >> 9,  k = c8 & (KC - 1);
    float  s   = c_sign[i * 8 + j];
    const float* src = w + (((size_t)c_widx[i * 8 + j] * NC + n) * KC + k);
    float4 a = *(const float4*)src;
    float4 b = *(const float4*)(src + 4);
    union { uint16_t h[8]; uint4 q; } o;
    o.h[0] = f2bf(s * a.x); o.h[1] = f2bf(s * a.y);
    o.h[2] = f2bf(s * a.z); o.h[3] = f2bf(s * a.w);
    o.h[4] = f2bf(s * b.x); o.h[5] = f2bf(s * b.y);
    o.h[6] = f2bf(s * b.z); o.h[7] = f2bf(s * b.w);
    ((uint4*)wb)[t] = o.q;
}

// ------------------------------ bf16 WMMA GEMM -----------------------------
// block = 256 thr = 8 waves; block tile 128(M) x 128(N);
// wave tile 64(M) x 32(N) = 4 x 2 WMMA accumulators; K step 32.
__global__ __launch_bounds__(256) void oct_gemm(const uint16_t* __restrict__ Xb,
                                                const uint16_t* __restrict__ Wb,
                                                float* __restrict__ Y) {
    const int lane = threadIdx.x & 31;
    const int wave = threadIdx.x >> 5;
    const int wm   = wave >> 2;          // 0..1
    const int wn   = wave & 3;           // 0..3
    const int m0   = blockIdx.y * 128 + wm * 64;
    const int n0   = blockIdx.x * 128 + wn * 32;
    const int lr   = lane & 15;          // row within 16-tile
    const int lh   = lane >> 4;          // lane half

    v8f acc[4][2] = {};

    // Per-lane base pointers.
    // A fragment (16x32 bf16): lane holds row m, k-chunks {8*lh..} and {16+8*lh..}
    // B fragment (32x16 bf16): lane holds col n (= Wbig row), k-chunks {8*lh..} and {16+8*lh..}
    const uint16_t* aPtr[4];
#pragma unroll
    for (int mi = 0; mi < 4; ++mi)
        aPtr[mi] = Xb + (size_t)(m0 + 16 * mi + lr) * DIN + 8 * lh;
    const uint16_t* bPtr[2];
#pragma unroll
    for (int ni = 0; ni < 2; ++ni)
        bPtr[ni] = Wb + (size_t)(n0 + 16 * ni + lr) * DIN + 8 * lh;

    union Fr { v16bf v; uint4 q[2]; };

    for (int k0 = 0; k0 < DIN; k0 += 32) {
        Fr a[4], b[2];
#pragma unroll
        for (int mi = 0; mi < 4; ++mi) {
            a[mi].q[0] = *(const uint4*)(aPtr[mi] + k0);
            a[mi].q[1] = *(const uint4*)(aPtr[mi] + k0 + 16);
        }
#pragma unroll
        for (int ni = 0; ni < 2; ++ni) {
            b[ni].q[0] = *(const uint4*)(bPtr[ni] + k0);
            b[ni].q[1] = *(const uint4*)(bPtr[ni] + k0 + 16);
        }
#pragma unroll
        for (int mi = 0; mi < 4; ++mi)
#pragma unroll
            for (int ni = 0; ni < 2; ++ni)
                acc[mi][ni] = __builtin_amdgcn_wmma_f32_16x16x32_bf16(
                    /*neg_a=*/false, a[mi].v,
                    /*neg_b=*/false, b[ni].v,
                    /*c_mod=*/(short)0, acc[mi][ni],
                    /*reuse_a=*/false, /*reuse_b=*/false);
    }

    // C/D layout: lane holds column n = lr, rows m = 8*lh + v (v = vgpr 0..7)
#pragma unroll
    for (int mi = 0; mi < 4; ++mi) {
#pragma unroll
        for (int ni = 0; ni < 2; ++ni) {
            int rb = m0 + 16 * mi + 8 * lh;
            int cc = n0 + 16 * ni + lr;
#pragma unroll
            for (int v = 0; v < 8; ++v)
                Y[(size_t)(rb + v) * DOUT + cc] = acc[mi][ni][v];
        }
    }
}

// ---------------------------------------------------------------------------
extern "C" void kernel_launch(void* const* d_in, const int* in_sizes, int n_in,
                              void* d_out, int out_size, void* d_ws, size_t ws_size,
                              hipStream_t stream) {
    const float* x = (const float*)d_in[0];   // [4,2048,4096] fp32
    const float* w = (const float*)d_in[1];   // [8,512,512]  fp32
    float* y = (float*)d_out;                  // [4,2048,4096] fp32

    uint16_t* xb = (uint16_t*)d_ws;                        // 64 MB bf16 X
    uint16_t* wb = xb + (size_t)BT * DIN;                  // 32 MB bf16 Wbig

    // 1) convert X to bf16
    oct_cvt_x<<<(BT * (size_t)DIN / 8) / 256, 256, 0, stream>>>(x, xb);
    // 2) build signed/gathered effective weight in bf16
    oct_build_w<<<((size_t)DOUT * DIN / 8) / 256, 256, 0, stream>>>(w, wb);
    // 3) WMMA GEMM: grid = (N tiles, M tiles) of 128x128
    dim3 grid(DOUT / 128, BT / 128);
    oct_gemm<<<grid, 256, 0, stream>>>(xb, wb, y);
}
